// SelfAttention_29575144800725
// MI455X (gfx1250) — matrix operand order repaired and checked by reading, not compiled, not run on hardware
//
#include <hip/hip_runtime.h>
#include <stdint.h>
#include <math.h>

typedef __attribute__((ext_vector_type(16))) _Float16 v16h;
typedef __attribute__((ext_vector_type(8)))  _Float16 v8h;
typedef __attribute__((ext_vector_type(8)))  float    v8f;
typedef __attribute__((ext_vector_type(4)))  uint32_t tdm_u4;
typedef __attribute__((ext_vector_type(8)))  int      tdm_i8;
typedef __attribute__((ext_vector_type(4)))  int      tdm_i4;

#define BDIM 4
#define SEQ  2048
#define DIM  1024
#define BM   128     // block rows (8 waves x 16)
#define BN   128     // block cols
#define KC   64      // K chunk staged per TDM transfer

// ---------------------------------------------------------------------------
// Issue one TDM 2D tile load: 128 rows x KC halfs, row stride = stride_elems.
// D# per CDNA5 ISA.  Tracked by TENSORcnt.
// ---------------------------------------------------------------------------
__device__ __forceinline__ void tdm_load_tile_f16(uint32_t lds_off,
                                                  const _Float16* gptr,
                                                  int stride_elems)
{
    const uint64_t ga = (uint64_t)(uintptr_t)gptr;

    tdm_u4 g0;
    g0[0] = 1u;                                   // count=1, user descriptor
    g0[1] = lds_off;                              // lds_addr (bytes)
    g0[2] = (uint32_t)ga;                         // global_addr[31:0]
    g0[3] = (uint32_t)((ga >> 32) & 0x01FFFFFFu)  // global_addr[56:32]
          | (2u << 30);                           // type = 2 ("image")

    uint32_t w0 = 1u << 16;                       // data_size = 1 -> 2 bytes
    // tensor_dim0 (bits 79:48) = KC ; tensor_dim1 (bits 111:80) = 128
    uint32_t w1 = ((uint32_t)KC & 0xFFFFu) << 16;
    uint32_t w2 = (((uint32_t)KC >> 16) & 0xFFFFu) | ((128u & 0xFFFFu) << 16);
    // tile_dim0 (bits 127:112) = KC
    uint32_t w3 = ((128u >> 16) & 0xFFFFu) | (((uint32_t)KC & 0xFFFFu) << 16);
    uint32_t w4 = 128u;                           // tile_dim1 = 128
    uint32_t w5 = (uint32_t)stride_elems;         // tensor_dim0_stride low 32
    uint32_t w6 = 0u, w7 = 0u;                    // stride hi / dim1_stride = 0 (2D)

    tdm_i8 g1;
    g1[0] = (int)w0; g1[1] = (int)w1; g1[2] = (int)w2; g1[3] = (int)w3;
    g1[4] = (int)w4; g1[5] = (int)w5; g1[6] = (int)w6; g1[7] = (int)w7;
    tdm_i4 z4 = {0, 0, 0, 0};

#if defined(__clang_major__) && (__clang_major__ >= 23)
    tdm_i8 z8 = {0, 0, 0, 0, 0, 0, 0, 0};
    __builtin_amdgcn_tensor_load_to_lds(g0, g1, z4, z4, z8, 0);
#else
    __builtin_amdgcn_tensor_load_to_lds(g0, g1, z4, z4, 0);
#endif
}

// ---------------------------------------------------------------------------
// Fragment loaders from LDS tiles (row-major [128][KC]); lane half h gets k = 16h..16h+15 of the k-step.
// ---------------------------------------------------------------------------
__device__ __forceinline__ v16h lds_a_frag(const _Float16* As,
                                           int row0, int ks, int lane) {
    const int r  = row0 + (lane & 15);
    const int kb = ks + ((lane >> 4) << 4);
    return *(const v16h*)(As + r * KC + kb);
}

__device__ __forceinline__ v16h lds_b_frag(const _Float16* Bs,
                                           int n0, int ks, int lane) {
    const int n  = n0 + (lane & 15);
    const int kb = ks + ((lane >> 4) << 4);
    return *(const v16h*)(Bs + n * KC + kb);
}

// ---------------------------------------------------------------------------
// NT WMMA GEMM with TDM-staged, double-buffered LDS tiles.
//   C[m,n] = alpha * sum_k A[m,k] * Bt[n,k]
// Block: 256 thr = 8 waves; block tile 128M x 128N; wave tile 16M x 128N.
// grid = (N/128, M/128, batch).  mode: 0 f32 out, 1 f16 out, 2 f16 transposed.
// LDS map (64 KB): [0] A buf0, [1] A buf1, [2] B buf0, [3] B buf1 (16 KB each)
// ---------------------------------------------------------------------------
__global__ __launch_bounds__(256)
void wmma_gemm_nt_tdm(const _Float16* __restrict__ A,  int lda, long long sA,
                      const _Float16* __restrict__ Bt, int ldb, long long sB,
                      void* __restrict__ Cout, int ldc, long long sC,
                      int K, float alpha, int mode)
{
    __shared__ __align__(128) _Float16 lds_all[4 * BM * KC];   // 64 KB

    const int lane = threadIdx.x & 31;
    const int wave = threadIdx.x >> 5;
    const int m0   = blockIdx.y * BM;
    const int n0   = blockIdx.x * BN;
    const int b    = blockIdx.z;

    A  += (size_t)b * (size_t)sA + (size_t)m0 * (size_t)lda;
    Bt += (size_t)b * (size_t)sB + (size_t)n0 * (size_t)ldb;

    // LDS byte offset: per ISA 10.2 the LDS aperture maps flat addr[31:0]
    // directly to the LDS byte offset (runtime cast, not a static init).
    const uint32_t lds_base = (uint32_t)(uintptr_t)(void*)&lds_all[0];
    const uint32_t BUFB     = (uint32_t)(BM * KC * 2);    // bytes per buffer

    const int NC = K / KC;

    // Preload chunks 0 and 1 (A+B each => 2 TDM ops per chunk, in-order).
    if (threadIdx.x < 32) {
        tdm_load_tile_f16(lds_base + 0u * BUFB, A,  lda);
        tdm_load_tile_f16(lds_base + 2u * BUFB, Bt, ldb);
        if (NC > 1) {
            tdm_load_tile_f16(lds_base + 1u * BUFB, A + KC,  lda);
            tdm_load_tile_f16(lds_base + 3u * BUFB, Bt + KC, ldb);
            __builtin_amdgcn_s_wait_tensorcnt((short)2);   // chunk 0 landed
        } else {
            __builtin_amdgcn_s_wait_tensorcnt((short)0);
        }
    }
    __syncthreads();

    v8f acc[8] = {};

    for (int c = 0; c < NC; ++c) {
        const int buf = c & 1;
        const _Float16* a_s = lds_all + (size_t)buf * (BM * KC);
        const _Float16* b_s = lds_all + (size_t)(2 + buf) * (BM * KC);

#pragma unroll
        for (int ks = 0; ks < KC; ks += 32) {
            // Load ALL fragments for this k-step first so the 8 WMMAs can
            // issue back-to-back behind a single ds-count wait instead of
            // serializing each WMMA on its own 2-load pair.
            const v16h af = lds_a_frag(a_s, wave * 16, ks, lane);
            v16h bf[8];
#pragma unroll
            for (int j = 0; j < 8; ++j)
                bf[j] = lds_b_frag(b_s, j * 16, ks, lane);
#pragma unroll
            for (int j = 0; j < 8; ++j)
                acc[j] = __builtin_amdgcn_wmma_f32_16x16x32_f16(
                    false, af, false, bf[j], (short)0, acc[j], false, false);
        }

        __syncthreads();                         // everyone done with buf
        if (threadIdx.x < 32) {
            if (c + 2 < NC) {                    // refill buf with chunk c+2
                tdm_load_tile_f16(lds_base + (uint32_t)buf * BUFB,
                                  A + (size_t)(c + 2) * KC, lda);
                tdm_load_tile_f16(lds_base + (uint32_t)(2 + buf) * BUFB,
                                  Bt + (size_t)(c + 2) * KC, ldb);
                __builtin_amdgcn_s_wait_tensorcnt((short)2); // chunk c+1 landed
            } else {
                __builtin_amdgcn_s_wait_tensorcnt((short)0); // drain
            }
        }
        __syncthreads();
    }

    // C/D layout: VGPR v -> M = m0 + wave*16 + v + 8*(lane/16), N = n0+j*16+lane%16
    const int mr = m0 + wave * 16 + ((lane >> 4) << 3);
    const int nc = lane & 15;

    if (mode == 0) {
        float* C = (float*)Cout + (size_t)b * (size_t)sC;
#pragma unroll
        for (int j = 0; j < 8; ++j) {
            const int col = n0 + j * 16 + nc;
#pragma unroll
            for (int v = 0; v < 8; ++v)
                C[(size_t)(mr + v) * (size_t)ldc + col] = acc[j][v] * alpha;
        }
    } else if (mode == 1) {
        _Float16* C = (_Float16*)Cout + (size_t)b * (size_t)sC;
#pragma unroll
        for (int j = 0; j < 8; ++j) {
            const int col = n0 + j * 16 + nc;
#pragma unroll
            for (int v = 0; v < 8; ++v)
                C[(size_t)(mr + v) * (size_t)ldc + col] = (_Float16)(acc[j][v] * alpha);
        }
    } else {   // transposed f16 store: CT[col*ldc + row]
        _Float16* CT = (_Float16*)Cout;
#pragma unroll
        for (int j = 0; j < 8; ++j) {
            const int col = n0 + j * 16 + nc;
#pragma unroll
            for (int v = 0; v < 8; ++v)
                CT[(size_t)col * (size_t)ldc + (mr + v)] = (_Float16)(acc[j][v] * alpha);
        }
    }
}

// ---------------------------------------------------------------------------
__global__ __launch_bounds__(256)
void cast_f32_f16(const float* __restrict__ x, _Float16* __restrict__ y, long long n)
{
    long long i      = (long long)blockIdx.x * blockDim.x + threadIdx.x;
    long long stride = (long long)gridDim.x * blockDim.x;
    for (; i < n; i += stride) y[i] = (_Float16)x[i];
}

// Row softmax over 2048 f32 values; result written IN-PLACE as f16 over the
// first half of the row's storage (all reads complete before any write).
__global__ __launch_bounds__(256)
void softmax_rows_inplace(float* __restrict__ scores)
{
    __shared__ float red[256];
    const int t = threadIdx.x;
    float* row = scores + (size_t)blockIdx.x * SEQ;

    float v[8];
#pragma unroll
    for (int i = 0; i < 8; ++i) v[i] = row[t * 8 + i];

    float m = v[0];
#pragma unroll
    for (int i = 1; i < 8; ++i) m = fmaxf(m, v[i]);
    red[t] = m;
    __syncthreads();
    for (int s = 128; s > 0; s >>= 1) {
        if (t < s) red[t] = fmaxf(red[t], red[t + s]);
        __syncthreads();
    }
    m = red[0];
    __syncthreads();

    float sum = 0.f;
#pragma unroll
    for (int i = 0; i < 8; ++i) { v[i] = __expf(v[i] - m); sum += v[i]; }
    red[t] = sum;
    __syncthreads();
    for (int s = 128; s > 0; s >>= 1) {
        if (t < s) red[t] += red[t + s];
        __syncthreads();
    }
    const float inv = 1.0f / red[0];

    _Float16* h = (_Float16*)row;
#pragma unroll
    for (int i = 0; i < 8; ++i) h[t * 8 + i] = (_Float16)(v[i] * inv);
}

// ---------------------------------------------------------------------------
extern "C" void kernel_launch(void* const* d_in, const int* in_sizes, int n_in,
                              void* d_out, int out_size, void* d_ws, size_t ws_size,
                              hipStream_t stream)
{
    const float* q  = (const float*)d_in[0];
    const float* Wq = (const float*)d_in[1];
    const float* Wk = (const float*)d_in[2];
    const float* Wv = (const float*)d_in[3];

    const long long MQ = (long long)BDIM * SEQ;   // 8192 flattened rows
    const long long nq = MQ * DIM;                // 8,388,608
    const long long nw = (long long)DIM * DIM;    // 1,048,576

    char* ws = (char*)d_ws;
    _Float16* qh  = (_Float16*)ws; ws += (size_t)nq * 2;
    _Float16* wqh = (_Float16*)ws; ws += (size_t)nw * 2;
    _Float16* wkh = (_Float16*)ws; ws += (size_t)nw * 2;
    _Float16* wvh = (_Float16*)ws; ws += (size_t)nw * 2;
    _Float16* qp  = (_Float16*)ws; ws += (size_t)nq * 2;
    _Float16* kp  = (_Float16*)ws; ws += (size_t)nq * 2;
    _Float16* vpT = (_Float16*)ws; ws += (size_t)nq * 2;   // [DIM, B*S] transposed
    float*    sc  = (float*)ws;                            // [B, S, S] f32 -> f16 in place

    const dim3 blk(256);

    // Stage 1: precision cast
    cast_f32_f16<<<4096, blk, 0, stream>>>(q,  qh,  nq);
    cast_f32_f16<<<1024, blk, 0, stream>>>(Wq, wqh, nw);
    cast_f32_f16<<<1024, blk, 0, stream>>>(Wk, wkh, nw);
    cast_f32_f16<<<1024, blk, 0, stream>>>(Wv, wvh, nw);

    // Stage 2: projections  P = X @ W^T   (M=8192, N=1024, K=1024)
    const dim3 gp(DIM / BN, (unsigned)(MQ / BM), 1);
    wmma_gemm_nt_tdm<<<gp, blk, 0, stream>>>(qh, DIM, 0, wqh, DIM, 0, qp,  DIM,     0, DIM, 1.0f, 1);
    wmma_gemm_nt_tdm<<<gp, blk, 0, stream>>>(qh, DIM, 0, wkh, DIM, 0, kp,  DIM,     0, DIM, 1.0f, 1);
    wmma_gemm_nt_tdm<<<gp, blk, 0, stream>>>(qh, DIM, 0, wvh, DIM, 0, vpT, (int)MQ, 0, DIM, 1.0f, 2);

    // Stage 3: scores = (1/sqrt(D)) * qp @ kp^T   per batch (M=N=2048, K=1024)
    const dim3 gs(SEQ / BN, SEQ / BM, BDIM);
    wmma_gemm_nt_tdm<<<gs, blk, 0, stream>>>(qp, DIM, (long long)SEQ * DIM,
                                             kp, DIM, (long long)SEQ * DIM,
                                             sc, SEQ, (long long)SEQ * SEQ,
                                             DIM, 0.03125f, 0);

    // Stage 4: row softmax, f32 -> f16 in place (attn lda = 2*SEQ halfs)
    softmax_rows_inplace<<<BDIM * SEQ, blk, 0, stream>>>(sc);

    // Stage 5: out = attn @ vp  per batch (M=2048, N=1024, K=2048)
    const dim3 go(DIM / BN, SEQ / BM, BDIM);
    wmma_gemm_nt_tdm<<<go, blk, 0, stream>>>((const _Float16*)sc, 2 * SEQ, (long long)SEQ * 2 * SEQ,
                                             vpT, (int)MQ, (long long)SEQ,   // batch shifts K window
                                             d_out, DIM, (long long)SEQ * DIM,
                                             SEQ, 1.0f, 0);

    (void)in_sizes; (void)n_in; (void)out_size; (void)ws_size;
}
